// EdgeDecoder_82171314307435
// MI455X (gfx1250) — matrix operand order repaired and checked by reading, not compiled, not run on hardware
//
#include <hip/hip_runtime.h>

typedef __attribute__((ext_vector_type(16))) _Float16     v16h;
typedef __attribute__((ext_vector_type(8)))  float        v8f;
typedef __attribute__((ext_vector_type(4)))  float        f4v;
typedef __attribute__((ext_vector_type(4)))  unsigned int u4v;

#define HDIM    256
#define NTILES  16              // 16 N-tiles of 16 -> H = 256
#define KBLOCKS 16              // 16 K-blocks of 32 -> 2H = 512
#define NFRAGS  (KBLOCKS * NTILES)                  // 256 B fragments
#define PLANE_DWORDS (NFRAGS * 32 * 4)              // 32768 dwords = 128 KB
#define W1_SWZ_DWORDS (2 * PLANE_DWORDS)            // 65536 dwords = 256 KB

// ---------------------------------------------------------------------------
// Prep: swizzle W1 [512,256] f32 (row-major W1[k][n]) into f16 B-fragment
// layout, stored as two 16-byte "planes" per fragment for conflict-free
// ds_load_b128 (lane stride = 16 B):
//   plane p (0/1), fragment f = kb*16+nt, lane L, dword dw (0..3):
//     ws[p*PLANE_DWORDS + (f*32 + L)*4 + dw]
//   holds elements e = (p*4+dw)*2 and +1 of the fragment, where per ISA
//   7.12.2 (16-bit B 32x16): n = nt*16 + L%16, K(e) = kb*32 + (e/8)*16 + (L/16)*8 + e%8.
// ---------------------------------------------------------------------------
__global__ void prep_w1_kernel(const float* __restrict__ W1,
                               unsigned int* __restrict__ wsw)
{
    int idx = blockIdx.x * blockDim.x + threadIdx.x;   // 0 .. 65535
    if (idx >= W1_SWZ_DWORDS) return;
    int p    = idx / PLANE_DWORDS;        // plane 0/1
    int r    = idx - p * PLANE_DWORDS;
    int dw   = r & 3;
    int lane = (r >> 2) & 31;
    int f    = r >> 7;
    int kb   = f >> 4;
    int nt   = f & 15;
    int g    = lane >> 4;
    int n    = nt * 16 + (lane & 15);
    int d    = p * 4 + dw;                // fragment dword 0..7
    int k    = kb * 32 + p * 16 + g * 8 + 2 * dw;

    _Float16 lo = (_Float16)W1[(long long)k * HDIM + n];
    _Float16 hi = (_Float16)W1[(long long)(k + 1) * HDIM + n];
    unsigned short lob = __builtin_bit_cast(unsigned short, lo);
    unsigned short hib = __builtin_bit_cast(unsigned short, hi);
    wsw[idx] = ((unsigned int)hib << 16) | (unsigned int)lob;
}

// ---------------------------------------------------------------------------
// Load one A fragment (16x32 f16, M x K) for this lane directly from a
// gathered embedding row.  Per ISA 7.12.2 (16-bit A 16x32):
//   lane L: M = L%16, g = L/16; elems 0..7  <- K = kb*32 + g*8 + e
//                               elems 8..15 <- K = kb*32 + 16 + g*8 + (e-8)
// s already points at row + (kb&7)*32 + g*8 (within the 256-float source row).
// ---------------------------------------------------------------------------
__device__ __forceinline__ v16h load_a_frag(const float* __restrict__ s)
{
    f4v x0 = *(const f4v*)(s + 0);
    f4v x1 = *(const f4v*)(s + 4);
    f4v x2 = *(const f4v*)(s + 16);
    f4v x3 = *(const f4v*)(s + 20);
    v16h a;
    a[0]  = (_Float16)x0[0];  a[1]  = (_Float16)x0[1];
    a[2]  = (_Float16)x0[2];  a[3]  = (_Float16)x0[3];
    a[4]  = (_Float16)x1[0];  a[5]  = (_Float16)x1[1];
    a[6]  = (_Float16)x1[2];  a[7]  = (_Float16)x1[3];
    a[8]  = (_Float16)x2[0];  a[9]  = (_Float16)x2[1];
    a[10] = (_Float16)x2[2];  a[11] = (_Float16)x2[3];
    a[12] = (_Float16)x3[0];  a[13] = (_Float16)x3[1];
    a[14] = (_Float16)x3[2];  a[15] = (_Float16)x3[3];
    return a;
}

struct BPair { u4v lo, hi; };   // 32 bytes -> bit_cast to v16h

// ---------------------------------------------------------------------------
// Main kernel: one wave owns 32 edges (two M=16 sub-tiles).
// ---------------------------------------------------------------------------
__global__ __launch_bounds__(256) void edge_mlp_kernel(
    const float* __restrict__ zd, const float* __restrict__ zt,
    const int*   __restrict__ row, const int* __restrict__ col,
    const float* __restrict__ b1,  const float* __restrict__ W2,
    const float* __restrict__ b2,
    const unsigned int* __restrict__ w1s,
    float* __restrict__ out, int nEdges)
{
    extern __shared__ unsigned int lds[];   // 256 KB swizzled f16 W1

    // Cooperative stage of swizzled W1 into LDS (b128 copies).
    {
        const u4v* __restrict__ src = (const u4v*)w1s;
        u4v* dst = (u4v*)lds;
        for (int i = threadIdx.x; i < W1_SWZ_DWORDS / 4; i += blockDim.x)
            dst[i] = src[i];
    }
    __syncthreads();

    const int lane = threadIdx.x & 31;
    const int wave = threadIdx.x >> 5;
    const int m    = lane & 15;     // M row within a 16-edge sub-tile
    const int g    = lane >> 4;     // lane half
    const int gw   = blockIdx.x * (blockDim.x >> 5) + wave;
    const int nw   = gridDim.x * (blockDim.x >> 5);

    // Per-lane epilogue constants: this lane's N-columns are nt*16 + m.
    float b1v[NTILES], w2v[NTILES];
#pragma unroll
    for (int nt = 0; nt < NTILES; ++nt) {
        b1v[nt] = b1[nt * 16 + m];
        w2v[nt] = W2[nt * 16 + m];
    }
    const float b2s = b2[0];

    // Two conflict-free B planes, lane stride 16 B.
    const u4v* __restrict__ bp0 = (const u4v*)lds;
    const u4v* __restrict__ bp1 = (const u4v*)(lds + PLANE_DWORDS);
    const int nMacro = nEdges >> 5;         // 32 edges per macro tile (exact)

    for (int t = gw; t < nMacro; t += nw) {
        const int base = t << 5;
        const int e0 = base + m;
        const int e1 = base + 16 + m;
        const float* pd0 = zd + (long long)row[e0] * HDIM;
        const float* pt0 = zt + (long long)col[e0] * HDIM;
        const float* pd1 = zd + (long long)row[e1] * HDIM;
        const float* pt1 = zt + (long long)col[e1] * HDIM;

        v8f acc0[NTILES], acc1[NTILES];
        const v8f vzero = {};
#pragma unroll
        for (int nt = 0; nt < NTILES; ++nt) { acc0[nt] = vzero; acc1[nt] = vzero; }

        for (int kb = 0; kb < KBLOCKS; ++kb) {
            const int off = ((kb & 7) << 5) + (g << 3);
            const float* s0 = (kb < 8 ? pd0 : pt0) + off;
            const float* s1 = (kb < 8 ? pd1 : pt1) + off;
            const v16h a0 = load_a_frag(s0);
            const v16h a1 = load_a_frag(s1);
            const int fbase = (kb << 4) * 32 + lane;     // u4v index of frag 0
#pragma unroll
            for (int nt = 0; nt < NTILES; ++nt) {
                BPair bp;
                bp.lo = bp0[fbase + nt * 32];
                bp.hi = bp1[fbase + nt * 32];
                const v16h bf = __builtin_bit_cast(v16h, bp);
                acc0[nt] = __builtin_amdgcn_wmma_f32_16x16x32_f16(
                    false, a0, false, bf, (short)0, acc0[nt], false, false);
                acc1[nt] = __builtin_amdgcn_wmma_f32_16x16x32_f16(
                    false, a1, false, bf, (short)0, acc1[nt], false, false);
            }
        }

        // Epilogue: h = relu(acc + b1); out[m] = h . W2 + b2.
        // D layout: lane = N%16 (half g), VGPR v = M row (v + 8*g).
#pragma unroll
        for (int s = 0; s < 2; ++s) {
            float p[8];
#pragma unroll
            for (int v = 0; v < 8; ++v) p[v] = 0.0f;
#pragma unroll
            for (int nt = 0; nt < NTILES; ++nt) {
                const v8f a = (s == 0) ? acc0[nt] : acc1[nt];
#pragma unroll
                for (int v = 0; v < 8; ++v) {
                    float hv = a[v] + b1v[nt];
                    hv = hv > 0.0f ? hv : 0.0f;
                    p[v] += hv * w2v[nt];
                }
            }
            // Reduce across the 16 lanes sharing one lane-half (sum over N).
#pragma unroll
            for (int v = 0; v < 8; ++v) {
#pragma unroll
                for (int msk = 1; msk < 16; msk <<= 1)
                    p[v] += __shfl_xor(p[v], msk, 32);
            }
            if (m == 0) {
#pragma unroll
                for (int v = 0; v < 8; ++v)
                    out[base + (s << 4) + (g << 3) + v] = p[v] + b2s;
            }
        }
    }
}

// ---------------------------------------------------------------------------
extern "C" void kernel_launch(void* const* d_in, const int* in_sizes, int n_in,
                              void* d_out, int out_size, void* d_ws, size_t ws_size,
                              hipStream_t stream)
{
    const float* zd  = (const float*)d_in[0];   // z_drug   [50000,256]
    const float* zt  = (const float*)d_in[1];   // z_target [20000,256]
    const int*   row = (const int*)  d_in[2];   // [E]
    const int*   col = (const int*)  d_in[3];   // [E]
    const float* W1  = (const float*)d_in[4];   // [512,256]
    const float* b1  = (const float*)d_in[5];   // [256]
    const float* W2  = (const float*)d_in[6];   // [256,1]
    const float* b2  = (const float*)d_in[7];   // [1]
    float* out = (float*)d_out;                 // [E,1] f32
    unsigned int* w1s = (unsigned int*)d_ws;    // 256 KB swizzled f16 W1
    const int nEdges = in_sizes[2];

    hipLaunchKernelGGL(prep_w1_kernel, dim3(W1_SWZ_DWORDS / 256), dim3(256),
                       0, stream, W1, w1s);
    hipLaunchKernelGGL(edge_mlp_kernel, dim3(256), dim3(256),
                       W1_SWZ_DWORDS * 4 /* 256 KB dynamic LDS */, stream,
                       zd, zt, row, col, b1, W2, b2, w1s, out, nEdges);
}
